// TriPlaneNetwork_55877524521416
// MI455X (gfx1250) — matrix operand order repaired and checked by reading, not compiled, not run on hardware
//
#include <hip/hip_runtime.h>

// ---------------------------------------------------------------------------
// TriPlaneNetwork for MI455X (gfx1250, wave32, WMMA)
//   Phase 1: tri-plane bilinear sampling -> canonical f16 [N][32] (in d_ws)
//   Phase 2: MLP chain via v_wmma_f32_16x16x32_f16, weights pre-swizzled in
//            LDS into exact B-fragment layout, activations staged per-wave.
// ---------------------------------------------------------------------------

typedef __attribute__((ext_vector_type(16))) _Float16 v16h;
typedef __attribute__((ext_vector_type(8)))  float    v8f;

#define NW     8      // waves per block (256 threads)
#define NFRAG  38     // total B fragments across all layers

__device__ __forceinline__ v8f wmma16(v16h a, v16h b, v8f c) {
  // D = A(16x32 f16) * B(32x16 f16) + C(16x16 f32)
  return __builtin_amdgcn_wmma_f32_16x16x32_f16(false, a, false, b, (short)0, c,
                                                false, false);
}

__device__ __forceinline__ float sigmoidf_(float x) {
  return 1.0f / (1.0f + __expf(-x));
}

// A-matrix 16x32 f16 fragment (ISA 7.12.2 layout):
//   lanes 0-15 : row = lane,    halves 0..7 = K 0..7,  halves 8..15 = K 16..23
//   lanes16-31 : row = lane-16, halves 0..7 = K 8..15, halves 8..15 = K 24..31
__device__ __forceinline__ v16h frag_A(const _Float16* base, int stride, int lane) {
  int r  = lane & 15;
  int kb = (lane >> 4) << 3;
  const _Float16* p = base + r * stride + kb;
  v16h a;
#pragma unroll
  for (int i = 0; i < 8; ++i) { a[i] = p[i]; a[i + 8] = p[i + 16]; }
  return a;
}

// B fragment: pre-swizzled in LDS, 512 halfs per fragment, 16 contiguous
// halfs per lane -> two ds_load_b128 per fragment.
__device__ __forceinline__ v16h frag_B(const _Float16* wswz, int f, int lane) {
  return *(const v16h*)(wswz + (f << 9) + (lane << 4));
}

// ---------------------------------------------------------------------------
// Phase 1: sampling.  thread = (t, n).  Channels 24..31 only (lo:hi = 24:32).
// canonical[n][t*8 + c] = sum_s prod_ci bilinear(planes_s[ci, 24+c], pts)
// ---------------------------------------------------------------------------
__global__ void __launch_bounds__(256) sample_kernel(
    const float* __restrict__ coords,   // [T][N][3]
    const float* __restrict__ p0, const float* __restrict__ p1,
    const float* __restrict__ p2, const float* __restrict__ p3,
    _Float16* __restrict__ canon,       // [N][32] f16
    int N)
{
  int n = blockIdx.x * blockDim.x + threadIdx.x;
  int t = blockIdx.y;
  if (n >= N) return;

  const float* c = coords + ((size_t)t * N + n) * 3;
  float px = c[0], py = c[1], pz = c[2];
  // COMBS = (0,1),(0,2),(1,2): x <- col a, y <- col b
  float xa[3] = {px, px, py};
  float yb[3] = {py, pz, pz};

  const float* planes[4] = {p0, p1, p2, p3};
  float feat[8] = {0.f, 0.f, 0.f, 0.f, 0.f, 0.f, 0.f, 0.f};

#pragma unroll
  for (int s = 0; s < 4; ++s) {
    const int R = 64 << s;
    const float* P = planes[s];
    const size_t psz = (size_t)R * R;     // one channel
    float prod[8] = {1.f, 1.f, 1.f, 1.f, 1.f, 1.f, 1.f, 1.f};
#pragma unroll
    for (int ci = 0; ci < 3; ++ci) {
      float fx = fminf(fmaxf((xa[ci] + 1.0f) * 0.5f * (float)(R - 1), 0.0f),
                       (float)(R - 1));
      float fy = fminf(fmaxf((yb[ci] + 1.0f) * 0.5f * (float)(R - 1), 0.0f),
                       (float)(R - 1));
      int x0 = (int)fx, y0 = (int)fy;
      int x1 = min(x0 + 1, R - 1), y1 = min(y0 + 1, R - 1);
      float wx = fx - (float)x0, wy = fy - (float)y0;
      const float* base = P + ((size_t)ci * 32 + 24) * psz;  // channel lo = 24
      size_t o00 = (size_t)y0 * R + x0, o01 = (size_t)y0 * R + x1;
      size_t o10 = (size_t)y1 * R + x0, o11 = (size_t)y1 * R + x1;
#pragma unroll
      for (int ch = 0; ch < 8; ++ch) {
        const float* g = base + (size_t)ch * psz;
        float v00 = g[o00], v01 = g[o01], v10 = g[o10], v11 = g[o11];
        float top = v00 * (1.0f - wx) + v01 * wx;
        float bot = v10 * (1.0f - wx) + v11 * wx;
        prod[ch] *= top * (1.0f - wy) + bot * wy;
      }
    }
#pragma unroll
    for (int ch = 0; ch < 8; ++ch) feat[ch] += prod[ch];
  }

  _Float16* o = canon + (size_t)n * 32 + t * 8;
#pragma unroll
  for (int ch = 0; ch < 8; ++ch) o[ch] = (_Float16)feat[ch];
}

// ---------------------------------------------------------------------------
// Phase 2: WMMA MLP.  One wave = one 16-point tile (grid-stride).
// Fragment table (B side), K padded by zero weights:
//   f 0..3  : L0  w_op0  [32][64]        1 kstep x 4 ntiles
//   f 4..11 : L1  w_op1  [64][64]        2 x 4
//   f12..17 : L2  w_op2  [64][33->48]    2 x 3
//   f18..25 : S0  w_sh0  [35->64][64]    2 x 4
//   f26..31 : S1  w_sh1  [64][48]        2 x 3
//   f32..37 : T   [w_dc|w_rest] [48->64][48]  2 x 3
// ---------------------------------------------------------------------------
__global__ void __launch_bounds__(256) mlp_kernel(
    const _Float16* __restrict__ canon, const float* __restrict__ dirs,
    const float* __restrict__ w0g, const float* __restrict__ w1g,
    const float* __restrict__ w2g, const float* __restrict__ ws0g,
    const float* __restrict__ ws1g, const float* __restrict__ wdcg,
    const float* __restrict__ bdcg, const float* __restrict__ wrestg,
    const float* __restrict__ brestg,
    float* __restrict__ out_op, float* __restrict__ out_dc,
    float* __restrict__ out_rest, int N)
{
  __shared__ __align__(32) _Float16 wswz[NFRAG * 512];  // 38 KB
  __shared__ __align__(32) _Float16 act[NW][16 * 64];   // 16 KB
  __shared__ float btail[48];

  // ---- cooperative swizzled weight load (B-fragment layout) ----
  for (int e = threadIdx.x; e < NFRAG * 512; e += blockDim.x) {
    int f    = e >> 9;
    int lane = (e >> 4) & 31;
    int i    = e & 15;
    int layer, nt, ks;
    if (f < 4)       { layer = 0; nt = f;            ks = 0;            }
    else if (f < 12) { layer = 1; nt = (f - 4) >> 1; ks = (f - 4) & 1;  }
    else if (f < 18) { layer = 2; nt = (f -12) >> 1; ks = (f -12) & 1;  }
    else if (f < 26) { layer = 3; nt = (f -18) >> 1; ks = (f -18) & 1;  }
    else if (f < 32) { layer = 4; nt = (f -26) >> 1; ks = (f -26) & 1;  }
    else             { layer = 5; nt = (f -32) >> 1; ks = (f -32) & 1;  }
    // B layout: lane l holds col (l&15), K = ks*32 + (l>>4)*8 + (i&7) + (i>>3)*16
    int n = nt * 16 + (lane & 15);
    int k = ks * 32 + ((lane >> 4) << 3) + (i & 7) + ((i >> 3) << 4);
    float v;
    switch (layer) {
      case 0:  v = w0g[k * 64 + n];                                     break;
      case 1:  v = w1g[k * 64 + n];                                     break;
      case 2:  v = (n < 33) ? w2g[k * 33 + n] : 0.0f;                   break;
      case 3:  v = (k < 35) ? ws0g[k * 64 + n] : 0.0f;                  break;
      case 4:  v = ws1g[k * 48 + n];                                    break;
      default: v = (k < 48) ? ((n < 3) ? wdcg[k * 3 + n]
                                       : wrestg[k * 45 + (n - 3)])
                            : 0.0f;                                     break;
    }
    wswz[e] = (_Float16)v;
  }
  if (threadIdx.x < 48)
    btail[threadIdx.x] = (threadIdx.x < 3) ? bdcg[threadIdx.x]
                                           : brestg[threadIdx.x - 3];
  __syncthreads();

  const int wave = threadIdx.x >> 5;
  const int lane = threadIdx.x & 31;
  _Float16* A = act[wave];
  const int nlo = lane & 15;           // C-fragment column
  const int m0  = (lane >> 4) << 3;    // C-fragment first row
  const int nTiles = N >> 4;           // N % 16 == 0 for this problem

  for (int tile = blockIdx.x * NW + wave; tile < nTiles;
       tile += gridDim.x * NW) {
    const int row0 = tile << 4;

    // ---- L0: canonical[16x32] @ w_op0 -> sigmoid -> act[16x64]
    v16h a0 = frag_A(canon + (size_t)row0 * 32, 32, lane);
#pragma unroll
    for (int nt = 0; nt < 4; ++nt) {
      v8f acc = {};
      acc = wmma16(a0, frag_B(wswz, nt, lane), acc);
#pragma unroll
      for (int v = 0; v < 8; ++v)
        A[(m0 + v) * 64 + nt * 16 + nlo] = (_Float16)sigmoidf_(acc[v]);
    }

    // ---- L1: act @ w_op1 -> sigmoid -> act  (A frags cached before overwrite)
    {
      v16h aa = frag_A(A, 64, lane);
      v16h ab = frag_A(A + 32, 64, lane);
#pragma unroll
      for (int nt = 0; nt < 4; ++nt) {
        v8f acc = {};
        acc = wmma16(aa, frag_B(wswz, 4 + nt * 2, lane), acc);
        acc = wmma16(ab, frag_B(wswz, 5 + nt * 2, lane), acc);
#pragma unroll
        for (int v = 0; v < 8; ++v)
          A[(m0 + v) * 64 + nt * 16 + nlo] = (_Float16)sigmoidf_(acc[v]);
      }
    }

    // ---- L2: act @ w_op2 -> opacity (col 0) + geo (cols 1..32 -> act 0..31)
    {
      v16h aa = frag_A(A, 64, lane);
      v16h ab = frag_A(A + 32, 64, lane);
#pragma unroll
      for (int nt = 0; nt < 3; ++nt) {
        v8f acc = {};
        acc = wmma16(aa, frag_B(wswz, 12 + nt * 2, lane), acc);
        acc = wmma16(ab, frag_B(wswz, 13 + nt * 2, lane), acc);
        int col = nt * 16 + nlo;
        if (col == 0) {
#pragma unroll
          for (int v = 0; v < 8; ++v)
            out_op[row0 + m0 + v] = sigmoidf_(acc[v]);
        } else if (col < 33) {
#pragma unroll
          for (int v = 0; v < 8; ++v)
            A[(m0 + v) * 64 + (col - 1)] = (_Float16)acc[v];
        }
      }
      // sh_in cols 32..34 = dirs; cols 35..63 are garbage but w_sh0 rows
      // 35..63 are zero-padded, so they never contribute.
      if (lane < 16) {
        const float* d = dirs + (size_t)(row0 + lane) * 3;
        A[lane * 64 + 32] = (_Float16)d[0];
        A[lane * 64 + 33] = (_Float16)d[1];
        A[lane * 64 + 34] = (_Float16)d[2];
      }
    }

    // ---- S0: sh_in @ w_sh0 (linear) -> act[16x64]
    {
      v16h aa = frag_A(A, 64, lane);
      v16h ab = frag_A(A + 32, 64, lane);
#pragma unroll
      for (int nt = 0; nt < 4; ++nt) {
        v8f acc = {};
        acc = wmma16(aa, frag_B(wswz, 18 + nt * 2, lane), acc);
        acc = wmma16(ab, frag_B(wswz, 19 + nt * 2, lane), acc);
#pragma unroll
        for (int v = 0; v < 8; ++v)
          A[(m0 + v) * 64 + nt * 16 + nlo] = (_Float16)acc[v];
      }
    }

    // ---- S1: @ w_sh1 (linear) -> act cols 0..47 (48..63 killed by zero pad)
    {
      v16h aa = frag_A(A, 64, lane);
      v16h ab = frag_A(A + 32, 64, lane);
#pragma unroll
      for (int nt = 0; nt < 3; ++nt) {
        v8f acc = {};
        acc = wmma16(aa, frag_B(wswz, 26 + nt * 2, lane), acc);
        acc = wmma16(ab, frag_B(wswz, 27 + nt * 2, lane), acc);
#pragma unroll
        for (int v = 0; v < 8; ++v)
          A[(m0 + v) * 64 + nt * 16 + nlo] = (_Float16)acc[v];
      }
    }

    // ---- tail: features_sh @ [w_dc|w_rest] + bias -> global outputs
    {
      v16h aa = frag_A(A, 64, lane);
      v16h ab = frag_A(A + 32, 64, lane);
#pragma unroll
      for (int nt = 0; nt < 3; ++nt) {
        v8f acc = {};
        acc = wmma16(aa, frag_B(wswz, 32 + nt * 2, lane), acc);
        acc = wmma16(ab, frag_B(wswz, 33 + nt * 2, lane), acc);
        int col = nt * 16 + nlo;
        float b = btail[col];
        if (col < 3) {
#pragma unroll
          for (int v = 0; v < 8; ++v)
            out_dc[(size_t)(row0 + m0 + v) * 3 + col] = acc[v] + b;
        } else {
#pragma unroll
          for (int v = 0; v < 8; ++v)
            out_rest[(size_t)(row0 + m0 + v) * 45 + (col - 3)] = acc[v] + b;
        }
      }
    }
  }
}

// ---------------------------------------------------------------------------
extern "C" void kernel_launch(void* const* d_in, const int* in_sizes, int n_in,
                              void* d_out, int out_size, void* d_ws,
                              size_t ws_size, hipStream_t stream) {
  const float* coords = (const float*)d_in[0];   // [4][N][3]
  const float* dirs   = (const float*)d_in[1];   // [N][3]
  const float* p0     = (const float*)d_in[2];
  const float* p1     = (const float*)d_in[3];
  const float* p2     = (const float*)d_in[4];
  const float* p3     = (const float*)d_in[5];
  const float* w_op0  = (const float*)d_in[6];
  const float* w_op1  = (const float*)d_in[7];
  const float* w_op2  = (const float*)d_in[8];
  const float* w_sh0  = (const float*)d_in[9];
  const float* w_sh1  = (const float*)d_in[10];
  const float* w_dc   = (const float*)d_in[11];
  const float* b_dc   = (const float*)d_in[12];
  const float* w_rest = (const float*)d_in[13];
  const float* b_rest = (const float*)d_in[14];

  const int N = in_sizes[1] / 3;                  // 500000
  const int T = in_sizes[0] / (3 * N);            // 4

  _Float16* canon = (_Float16*)d_ws;              // [N][32] f16 = 32 MB

  float* out_op   = (float*)d_out;                // [N]
  float* out_dc   = out_op + N;                   // [N][3]
  float* out_rest = out_op + 4 * (size_t)N;       // [N][45]

  dim3 g1((N + 255) / 256, T);
  sample_kernel<<<g1, 256, 0, stream>>>(coords, p0, p1, p2, p3, canon, N);

  int nTiles = N / 16;
  int blocks = (nTiles + NW - 1) / NW;
  if (blocks > 1024) blocks = 1024;               // persistent grid-stride
  mlp_kernel<<<blocks, 256, 0, stream>>>(canon, dirs, w_op0, w_op1, w_op2,
                                         w_sh0, w_sh1, w_dc, b_dc, w_rest,
                                         b_rest, out_op, out_dc, out_rest, N);
}
